// Gemma3nTextAttention_87789131530457
// MI455X (gfx1250) — matrix operand order repaired and checked by reading, not compiled, not run on hardware
//
#include <hip/hip_runtime.h>
#include <hip/hip_bf16.h>

// ---------------------------------------------------------------------------
// Gemma3n attention for MI455X (gfx1250, wave32, WMMA).
// B=2, S=2048, HID=2048, NH=8, NKV=2, HD=256, WINDOW=512.
// Strategy: prepack fp32 -> bf16 once; all GEMMs and attention matmuls run
// pure bf16 loads + v_wmma_f32_16x16x32_bf16 with fp32 accumulation.
// ---------------------------------------------------------------------------

#define Bz   2
#define Sz   2048
#define HIDz 2048
#define NHz  8
#define NKVz 2
#define HDz  256
#define WINz 512
#define Mz   (Bz * Sz)          // 4096 token rows

typedef __attribute__((ext_vector_type(16))) __bf16          v16bf;
typedef __attribute__((ext_vector_type(8)))  float           v8f;
typedef __attribute__((ext_vector_type(4)))  unsigned short  u16x4;
typedef __attribute__((ext_vector_type(8)))  unsigned short  u16x8;
typedef __attribute__((ext_vector_type(16))) unsigned short  u16x16;

static __device__ inline v8f wmma_bf16(v16bf a, v16bf b, v8f c) {
  // D = A(16x32 bf16) x B(32x16 bf16) + C(16x16 f32)
  return __builtin_amdgcn_wmma_f32_16x16x32_bf16(
      /*neg_a=*/false, a, /*neg_b=*/false, b,
      /*c_mod=*/(short)0, c, /*reuse_a=*/false, /*reuse_b=*/false);
}

// float -> bf16 bits, round-to-nearest-even
static __device__ inline unsigned short f2bfu(float f) {
  unsigned u = __builtin_bit_cast(unsigned, f);
  u += 0x7FFFu + ((u >> 16) & 1u);
  return (unsigned short)(u >> 16);
}

// B operand: 16 contiguous bf16 halves (lane n holds K = 16*hi .. 16*hi+15)
static __device__ inline v16bf ldB16(const unsigned short* p) {
  return __builtin_bit_cast(v16bf, *(const u16x16*)p);
}

// A operand from bf16 memory: row-chunk base pointer; lane-half hi picks the
// contiguous K chunk {16hi..16hi+15}.
static __device__ inline v16bf ldA16(const unsigned short* row, int hi) {
  return __builtin_bit_cast(v16bf, *(const u16x16*)(row + 16 * hi));
}

static __device__ inline v8f v8f_zero() {
  v8f z = {0.f, 0.f, 0.f, 0.f, 0.f, 0.f, 0.f, 0.f};
  return z;
}

// ---------------------------------------------------------------------------
// Kernel 0: one-pass fp32 -> bf16 prepack (n must be a multiple of 4).
// ---------------------------------------------------------------------------
__global__ __launch_bounds__(256) void cvt_bf16_kernel(
    const float* __restrict__ src, unsigned short* __restrict__ dst, int n) {
  const int i = (blockIdx.x * blockDim.x + threadIdx.x) * 4;
  if (i >= n) return;
  const float4 v = *(const float4*)(src + i);
  u16x4 r;
  r[0] = f2bfu(v.x); r[1] = f2bfu(v.y); r[2] = f2bfu(v.z); r[3] = f2bfu(v.w);
  *(u16x4*)(dst + i) = r;
}

// ---------------------------------------------------------------------------
// Kernel 1/4: Y[M,N] = A[M,K] * W[N,K]^T   (bf16 in, fp32 out)
// One wave computes a 64x64 tile: 16 f32 accumulators, 16 WMMA per K-step.
// Inner loop is pure bf16 loads + WMMA (no conversion math).
// ---------------------------------------------------------------------------
__global__ __launch_bounds__(256) void gemm_nt_bb(
    const unsigned short* __restrict__ A, const unsigned short* __restrict__ W,
    float* __restrict__ Y, int M, int N, int K) {
  const int lane = threadIdx.x & 31;
  const int wave = threadIdx.x >> 5;
  const int ntiles = N >> 6;
  const int mtiles = M >> 6;
  const int tile = blockIdx.x * (blockDim.x >> 5) + wave;
  if (tile >= mtiles * ntiles) return;   // uniform per wave: EXEC stays all-1s
  const int m0 = (tile / ntiles) << 6;
  const int n0 = (tile % ntiles) << 6;
  const int nl = lane & 15, hi = lane >> 4;

  v8f acc[4][4];
#pragma unroll
  for (int i = 0; i < 4; ++i)
#pragma unroll
    for (int j = 0; j < 4; ++j) acc[i][j] = v8f_zero();

  for (int k0 = 0; k0 < K; k0 += 32) {
    v16bf Bt[4];
#pragma unroll
    for (int nt = 0; nt < 4; ++nt)
      Bt[nt] = ldB16(W + (size_t)(n0 + nt * 16 + nl) * K + k0 + 16 * hi);
#pragma unroll
    for (int mt = 0; mt < 4; ++mt) {
      v16bf At = ldA16(A + (size_t)(m0 + mt * 16 + nl) * K + k0, hi);
#pragma unroll
      for (int nt = 0; nt < 4; ++nt)
        acc[mt][nt] = wmma_bf16(At, Bt[nt], acc[mt][nt]);
    }
  }

#pragma unroll
  for (int mt = 0; mt < 4; ++mt)
#pragma unroll
    for (int nt = 0; nt < 4; ++nt)
#pragma unroll
      for (int r = 0; r < 8; ++r) {
        int row = m0 + mt * 16 + r + 8 * hi;
        int col = n0 + nt * 16 + nl;
        Y[(size_t)row * N + col] = acc[mt][nt][r];
      }
}

// ---------------------------------------------------------------------------
// Kernel 2: per-head RMSNorm (+RoPE for q/k) and bf16 pack.
// One wave per 256-dim head vector; lane holds 8 elems d = j*32+lane.
// Outputs: qh[b,h,s,d], kh[b,kv,s,d] row-major bf16; vT[b,kv,d,s] transposed.
// ---------------------------------------------------------------------------
static __device__ inline void norm_rope_store(
    const float* __restrict__ src, const float* __restrict__ wv,
    const float* __restrict__ cosp, const float* __restrict__ sinp,
    bool rope, unsigned short* __restrict__ dst, int dstride, int lane) {
  float x[8];
#pragma unroll
  for (int j = 0; j < 8; ++j) x[j] = src[j * 32 + lane];
  float ss = 0.f;
#pragma unroll
  for (int j = 0; j < 8; ++j) ss += x[j] * x[j];
#pragma unroll
  for (int off = 16; off > 0; off >>= 1) ss += __shfl_xor(ss, off, 32);
  const float rs = rsqrtf(ss * (1.f / (float)HDz) + 1e-6f);
  float y[8];
#pragma unroll
  for (int j = 0; j < 8; ++j) {
    const int d = j * 32 + lane;
    const float w = wv ? wv[d] : 1.f;
    y[j] = x[j] * rs * w;
  }
  if (rope) {
    float o[8];
#pragma unroll
    for (int j = 0; j < 8; ++j) {
      const int d = j * 32 + lane;                  // d<128 <=> j<4
      const float rh = (j < 4) ? -y[j + 4] : y[j - 4];
      o[j] = y[j] * cosp[d] + rh * sinp[d];
    }
#pragma unroll
    for (int j = 0; j < 8; ++j) y[j] = o[j];
  }
#pragma unroll
  for (int j = 0; j < 8; ++j) {
    const int d = j * 32 + lane;
    dst[(size_t)d * dstride] = f2bfu(y[j]);
  }
}

__global__ __launch_bounds__(256) void normrope_kernel(
    const float* __restrict__ Qraw, const float* __restrict__ Kraw,
    const float* __restrict__ Vraw, const float* __restrict__ cosp,
    const float* __restrict__ sinp, const float* __restrict__ qw,
    const float* __restrict__ kw, unsigned short* __restrict__ qh,
    unsigned short* __restrict__ kh, unsigned short* __restrict__ vT) {
  const int wave = threadIdx.x >> 5, lane = threadIdx.x & 31;
  const int m = blockIdx.x;           // token row in [0, B*S)
  const int b = m >> 11, s = m & (Sz - 1);
  const float* cp = cosp + (size_t)m * HDz;
  const float* sp = sinp + (size_t)m * HDz;

  // q head = wave (8 heads / 8 waves)
  norm_rope_store(Qraw + (size_t)m * (NHz * HDz) + wave * HDz, qw, cp, sp, true,
                  qh + ((size_t)(b * NHz + wave) * Sz + s) * HDz, 1, lane);
  if (wave < 2) {              // k heads
    norm_rope_store(Kraw + (size_t)m * (NKVz * HDz) + wave * HDz, kw, cp, sp,
                    true, kh + ((size_t)(b * NKVz + wave) * Sz + s) * HDz, 1,
                    lane);
  } else if (wave < 4) {       // v heads (unit weight, no rope, transposed out)
    const int h = wave - 2;
    norm_rope_store(Vraw + (size_t)m * (NKVz * HDz) + h * HDz, nullptr, cp, sp,
                    false, vT + ((size_t)(b * NKVz + h) * HDz) * Sz + s, Sz,
                    lane);
  }
}

// ---------------------------------------------------------------------------
// Kernel 3: sliding-window flash attention. One wave = one 16-row Q tile.
// Per 32-key block: 16 WMMA (Q.K^T) + online softmax + 16 WMMA (P.V).
// Output written directly as bf16 for the final projection GEMM.
// ---------------------------------------------------------------------------
__global__ __launch_bounds__(128) void attn_kernel(
    const unsigned short* __restrict__ qh, const unsigned short* __restrict__ kh,
    const unsigned short* __restrict__ vT, unsigned short* __restrict__ attn) {
  __shared__ __align__(16) unsigned short pbuf[4 * 16 * 32];  // per-wave P tile
  const int lane = threadIdx.x & 31, wave = threadIdx.x >> 5;
  const int S16 = Sz / 16;  // 128 q-tiles per head
  const int tile = blockIdx.x * 4 + wave;
  if (tile >= Bz * NHz * S16) return;
  const int b = tile / (NHz * S16);
  const int rem = tile % (NHz * S16);
  const int h = rem / S16, qt = rem % S16;
  const int kvh = h >> 2;   // rep = NH/NKV = 4
  const int qi0 = qt * 16;
  const int nl = lane & 15, hi = lane >> 4;

  const unsigned short* qb = qh + ((size_t)(b * NHz + h) * Sz) * HDz;
  const unsigned short* kb = kh + ((size_t)(b * NKVz + kvh) * Sz) * HDz;
  const unsigned short* vb = vT + ((size_t)(b * NKVz + kvh) * HDz) * Sz;

  // Q tile resident in A-operand layout: 8 chunks of 32 dims
  v16bf Qa[8];
#pragma unroll
  for (int c = 0; c < 8; ++c)
    Qa[c] = ldA16(qb + (size_t)(qi0 + nl) * HDz + 32 * c, hi);

  v8f acc[16];
#pragma unroll
  for (int t = 0; t < 16; ++t) acc[t] = v8f_zero();
  float Mr[8], Lr[8];
#pragma unroll
  for (int r = 0; r < 8; ++r) { Mr[r] = -1e30f; Lr[r] = 0.f; }

  unsigned short* pb = pbuf + wave * (16 * 32);

  int kb0 = qi0 - (WINz - 1);
  if (kb0 < 0) kb0 = 0;
  kb0 &= ~31;

  for (int kblk = kb0; kblk <= qi0 + 15; kblk += 32) {
    // ---- scores: two 16x16 tiles over 32 keys, K = HD = 256 (8 steps)
    v8f s0 = v8f_zero(), s1 = v8f_zero();
#pragma unroll
    for (int c = 0; c < 8; ++c) {
      v16bf B0 = ldB16(kb + (size_t)(kblk + nl) * HDz + 32 * c + 16 * hi);
      v16bf B1 = ldB16(kb + (size_t)(kblk + 16 + nl) * HDz + 32 * c + 16 * hi);
      s0 = wmma_bf16(Qa[c], B0, s0);
      s1 = wmma_bf16(Qa[c], B1, s1);
    }
    // ---- online softmax (row m = r + 8*hi lives across 16 lanes of a half)
    float p0[8], p1[8], alpha[8];
#pragma unroll
    for (int r = 0; r < 8; ++r) {
      const int i = qi0 + r + 8 * hi;
      const int j0 = kblk + nl, j1 = j0 + 16;
      const bool ok0 = (j0 <= i) && (i - j0 < WINz);
      const bool ok1 = (j1 <= i) && (i - j1 < WINz);
      float mx = fmaxf(ok0 ? s0[r] : -1e30f, ok1 ? s1[r] : -1e30f);
#pragma unroll
      for (int off = 8; off > 0; off >>= 1) mx = fmaxf(mx, __shfl_xor(mx, off, 32));
      const float mnew = fmaxf(Mr[r], mx);
      alpha[r] = __expf(Mr[r] - mnew);
      p0[r] = ok0 ? __expf(s0[r] - mnew) : 0.f;
      p1[r] = ok1 ? __expf(s1[r] - mnew) : 0.f;
      float rsum = p0[r] + p1[r];
#pragma unroll
      for (int off = 8; off > 0; off >>= 1) rsum += __shfl_xor(rsum, off, 32);
      Lr[r] = Lr[r] * alpha[r] + rsum;
      Mr[r] = mnew;
    }
#pragma unroll
    for (int t = 0; t < 16; ++t)
#pragma unroll
      for (int r = 0; r < 8; ++r) acc[t][r] *= alpha[r];

    // ---- relayout P: C-layout -> A-layout via per-wave LDS (DS is in-order)
#pragma unroll
    for (int r = 0; r < 8; ++r) {
      const int mrow = r + 8 * hi;
      pb[mrow * 32 + nl]      = f2bfu(p0[r]);
      pb[mrow * 32 + 16 + nl] = f2bfu(p1[r]);
    }
    v16bf Pa = ldA16(pb + nl * 32, hi);

    // ---- O += P(16x32) x V(32x256): 16 N-tiles, B from transposed vT
#pragma unroll
    for (int t = 0; t < 16; ++t) {
      v16bf Bv = ldB16(vb + (size_t)(16 * t + nl) * Sz + kblk + 16 * hi);
      acc[t] = wmma_bf16(Pa, Bv, acc[t]);
    }
  }

  // ---- normalize and store bf16 [B*S, NH*HD] for the output projection
#pragma unroll
  for (int r = 0; r < 8; ++r) {
    const float inv = 1.f / Lr[r];
    const int mrow = r + 8 * hi;
    const size_t row = (size_t)(b * Sz + qi0 + mrow) * (NHz * HDz) + h * HDz;
#pragma unroll
    for (int t = 0; t < 16; ++t)
      attn[row + t * 16 + nl] = f2bfu(acc[t][r] * inv);
  }
}

// ---------------------------------------------------------------------------
extern "C" void kernel_launch(void* const* d_in, const int* in_sizes, int n_in,
                              void* d_out, int out_size, void* d_ws,
                              size_t ws_size, hipStream_t stream) {
  const float* hs  = (const float*)d_in[0];   // [B,S,HID]
  const float* cp  = (const float*)d_in[1];   // [B,S,HD]
  const float* sp  = (const float*)d_in[2];   // [B,S,HD]
  const float* Wq  = (const float*)d_in[3];   // [NH*HD, HID]
  const float* Wk  = (const float*)d_in[4];   // [NKV*HD, HID]
  const float* Wv  = (const float*)d_in[5];   // [NKV*HD, HID]
  const float* Wo  = (const float*)d_in[6];   // [HID, HID]
  const float* qw  = (const float*)d_in[7];   // [HD]
  const float* kw  = (const float*)d_in[8];   // [HD]
  float* out = (float*)d_out;                 // [B,S,HID]

  char* ws = (char*)d_ws;
  size_t off = 0;
  unsigned short* hs_bf = (unsigned short*)(ws + off); off += (size_t)Mz * HIDz * 2;          // 16MB
  unsigned short* Wq_bf = (unsigned short*)(ws + off); off += (size_t)NHz * HDz * HIDz * 2;   //  8MB
  unsigned short* Wk_bf = (unsigned short*)(ws + off); off += (size_t)NKVz * HDz * HIDz * 2;  //  2MB
  unsigned short* Wv_bf = (unsigned short*)(ws + off); off += (size_t)NKVz * HDz * HIDz * 2;  //  2MB
  unsigned short* Wo_bf = (unsigned short*)(ws + off); off += (size_t)HIDz * HIDz * 2;        //  8MB
  float* Qraw = (float*)(ws + off); off += (size_t)Mz * (NHz * HDz) * 4;   // 32MB (reused as attn_bf)
  float* Kraw = (float*)(ws + off); off += (size_t)Mz * (NKVz * HDz) * 4;  //  8MB
  float* Vraw = (float*)(ws + off); off += (size_t)Mz * (NKVz * HDz) * 4;  //  8MB
  unsigned short* qh = (unsigned short*)(ws + off); off += (size_t)Mz * (NHz * HDz) * 2;   // 16MB
  unsigned short* kh = (unsigned short*)(ws + off); off += (size_t)Mz * (NKVz * HDz) * 2;  //  4MB
  unsigned short* vT = (unsigned short*)(ws + off); off += (size_t)Mz * (NKVz * HDz) * 2;  //  4MB
  unsigned short* attn_bf = (unsigned short*)Qraw;  // Qraw dead after normrope

  auto cblk = [](int n) { return (n / 4 + 255) / 256; };
  auto gblk = [](int mt, int nt) { return (mt * nt + 7) / 8; };

  // ---- prepack fp32 -> bf16 (one pass each)
  cvt_bf16_kernel<<<cblk(Mz * HIDz), 256, 0, stream>>>(hs, hs_bf, Mz * HIDz);
  cvt_bf16_kernel<<<cblk(NHz * HDz * HIDz), 256, 0, stream>>>(Wq, Wq_bf, NHz * HDz * HIDz);
  cvt_bf16_kernel<<<cblk(NKVz * HDz * HIDz), 256, 0, stream>>>(Wk, Wk_bf, NKVz * HDz * HIDz);
  cvt_bf16_kernel<<<cblk(NKVz * HDz * HIDz), 256, 0, stream>>>(Wv, Wv_bf, NKVz * HDz * HIDz);
  cvt_bf16_kernel<<<cblk(HIDz * HIDz), 256, 0, stream>>>(Wo, Wo_bf, HIDz * HIDz);

  // ---- Q/K/V projections (bf16 in, fp32 out)
  gemm_nt_bb<<<gblk(Mz / 64, 2048 / 64), 256, 0, stream>>>(hs_bf, Wq_bf, Qraw, Mz, 2048, HIDz);
  gemm_nt_bb<<<gblk(Mz / 64,  512 / 64), 256, 0, stream>>>(hs_bf, Wk_bf, Kraw, Mz,  512, HIDz);
  gemm_nt_bb<<<gblk(Mz / 64,  512 / 64), 256, 0, stream>>>(hs_bf, Wv_bf, Vraw, Mz,  512, HIDz);

  // ---- RMSNorm + RoPE + bf16 pack (V transposed)
  normrope_kernel<<<Mz, 256, 0, stream>>>(Qraw, Kraw, Vraw, cp, sp, qw, kw, qh, kh, vT);

  // ---- Sliding-window attention: 2048 wave-tiles, 4 waves/block
  attn_kernel<<<(Bz * NHz * (Sz / 16)) / 4, 128, 0, stream>>>(qh, kh, vT, attn_bf);

  // ---- Output projection
  gemm_nt_bb<<<gblk(Mz / 64, 2048 / 64), 256, 0, stream>>>(attn_bf, Wo_bf, out, Mz, HIDz, HIDz);
}
